// Model_Inference_33938831573442
// MI455X (gfx1250) — matrix operand order repaired and checked by reading, not compiled, not run on hardware
//
#include <hip/hip_runtime.h>

// ---------------------------------------------------------------------------
// GNN inference for MI455X (gfx1250, wave32, WMMA).
// Heavy work = per-edge MLP GEMMs done with v_wmma_f32_16x16x32_f16.
// Persistent blocks: weights staged to LDS once per block, then each wave
// grid-strides over 16-edge tiles.
// ---------------------------------------------------------------------------

typedef __attribute__((ext_vector_type(16))) _Float16 v16h;
typedef __attribute__((ext_vector_type(8)))  float    v8f;

#define FDIM 16
#define DDIM 64
#define HDIM 64
#define CDIM 6

// ---------------------------------------------------------------- zero ------
__global__ void zero_kernel(float* __restrict__ p, long long n) {
    long long i = (long long)blockIdx.x * blockDim.x + threadIdx.x;
    long long stride = (long long)gridDim.x * blockDim.x;
    for (; i < n; i += stride) p[i] = 0.0f;
}

// ------------------------------------------------------------- degree -------
__global__ void degree_kernel(const long long* __restrict__ ei,
                              float* __restrict__ deg, int E) {
    int i = blockIdx.x * blockDim.x + threadIdx.x;
    if (i < E) {
        long long d = ei[(long long)E + i];   // edge_index[1][i] = dst
        atomicAdd(&deg[d], 1.0f);
    }
}

__global__ void invdeg_kernel(float* __restrict__ deg, int n) {
    int i = blockIdx.x * blockDim.x + threadIdx.x;
    if (i < n) deg[i] = 1.0f / fmaxf(deg[i], 1.0f);
}

// ------------------------------------------------------------- encoder ------
// x0 = relu(nf @ enc_W0 + enc_b0)   [N,16] @ [16,64]
__global__ void encoder_kernel(const float* __restrict__ nf,
                               const float* __restrict__ W,
                               const float* __restrict__ b,
                               float* __restrict__ x0, int n) {
    __shared__ float sW[FDIM * DDIM];
    __shared__ float sB[DDIM];
    for (int t = threadIdx.x; t < FDIM * DDIM; t += blockDim.x) sW[t] = W[t];
    if (threadIdx.x < DDIM) sB[threadIdx.x] = b[threadIdx.x];
    __syncthreads();
    int i = blockIdx.x * blockDim.x + threadIdx.x;
    if (i >= n) return;
    float in[FDIM];
#pragma unroll
    for (int k = 0; k < FDIM; ++k) in[k] = nf[(long long)i * FDIM + k];
#pragma unroll 4
    for (int o = 0; o < DDIM; ++o) {
        float acc = sB[o];
#pragma unroll
        for (int k = 0; k < FDIM; ++k) acc = fmaf(in[k], sW[k * DDIM + o], acc);
        x0[(long long)i * DDIM + o] = fmaxf(acc, 0.0f);
    }
}

// ------------------------------------------------------------ conv layer ----
// Per 16-edge tile (one wave32):
//   A = [x[dst] | x[src]-x[dst]]   (16 x 128, f16)
//   hid = relu(A @ Wm1 + bm1)      via 16 wmma  (4 ntile x 4 kchunk)
//   msg = relu(hid @ Wm2 + bm2)    via 8  wmma  (4 ntile x 2 kchunk)
//   x_out[dst] += msg * inv_deg[dst]   (f32 global atomics)
//
// A-matrix 16-bit lane layout (16xK chunk of 32):
//   lane<16 : M=lane,    K offsets {8g+j | j<8} U {16+8g+(j-8)} with g=0
//   lane>=16: M=lane-16, same with g=1
// B-matrix 16-bit lane layout (32x16): same K order as A, N=16nt+m
// C/D layout: VGPR r, lane => M = r + 8g, N = 16nt + m
__global__ __launch_bounds__(256) void conv_kernel(
    const float* __restrict__ x_in, float* __restrict__ x_out,
    const long long* __restrict__ ei, const float* __restrict__ inv_deg,
    const float* __restrict__ Wm1, const float* __restrict__ bm1,
    const float* __restrict__ Wm2, const float* __restrict__ bm2, int E) {

    // fragment-major weight staging: frag index f=(c*4+nt), per-lane 16 halves
    // contiguous => single 32B LDS read per B fragment.
    __shared__ __align__(32) _Float16 sW1f[16 * 32 * 16];   // 16 KB
    __shared__ __align__(32) _Float16 sW2f[8 * 32 * 16];    //  8 KB
    __shared__ _Float16 sHid[8][16 * HDIM];                 // 16 KB (per-wave)
    __shared__ float sB1[HDIM], sB2[DDIM];
    __shared__ int   sDst[8][16];
    __shared__ float sInv[8][16];

    const int tid = threadIdx.x;

    // stage Wm1 [k=128][n=64] into fragment-major f16 (once per block)
    for (int t = tid; t < 2 * DDIM * HDIM; t += 256) {
        int k = t >> 6, n = t & 63;
        int c = k >> 5, g2 = (k >> 3) & 1, j = (k & 7) + (((k >> 4) & 1) << 3);
        int nt = n >> 4, mm = n & 15;
        sW1f[(((c * 4 + nt) * 32) + (g2 * 16 + mm)) * 16 + j] = (_Float16)Wm1[t];
    }
    // stage Wm2 [k=64][n=64]
    for (int t = tid; t < HDIM * DDIM; t += 256) {
        int k = t >> 6, n = t & 63;
        int c = k >> 5, g2 = (k >> 3) & 1, j = (k & 7) + (((k >> 4) & 1) << 3);
        int nt = n >> 4, mm = n & 15;
        sW2f[(((c * 4 + nt) * 32) + (g2 * 16 + mm)) * 16 + j] = (_Float16)Wm2[t];
    }
    if (tid < HDIM) sB1[tid] = bm1[tid];
    if (tid >= 128 && tid < 128 + DDIM) sB2[tid - 128] = bm2[tid - 128];
    __syncthreads();   // weights visible to all waves

    const int wave = tid >> 5;
    const int lane = tid & 31;
    const int m    = lane & 15;
    const int g    = lane >> 4;
    const v8f vzero = {0, 0, 0, 0, 0, 0, 0, 0};

    const int tiles = (E + 15) / 16;
    const int tileStride = gridDim.x * 8;

    // Persistent loop: t0 is block-uniform, so all waves hit the same number
    // of barriers. Tail waves (tile >= tiles) clamp loads and skip stores.
    for (int t0 = blockIdx.x * 8; t0 < tiles; t0 += tileStride) {
        const int tile = t0 + wave;
        const int e0   = tile * 16;

        // gather this lane's edge row (lane and lane+16 share row m)
        int e = e0 + m; if (e >= E) e = E - 1; if (e < 0) e = 0;
        long long sIdx = ei[e];
        long long dIdx = ei[(long long)E + e];
        if (g == 0) {
            sDst[wave][m] = (int)dIdx;
            sInv[wave][m] = inv_deg[dIdx];
        }

        const float* xd_row = x_in + (long long)dIdx * DDIM;
        const float* xs_row = x_in + (long long)sIdx * DDIM;

        // A fragments, 4 K-chunks of 32 over 2D=128: chunks 0,1 = xd; 2,3 = xs-xd
        v16h a[4];
#pragma unroll
        for (int j = 0; j < 16; ++j) {
            int kk = (j < 8) ? (8 * g + j) : (16 + 8 * g + (j - 8));  // in [0,32)
            float xd0 = xd_row[kk],      xs0 = xs_row[kk];
            float xd1 = xd_row[32 + kk], xs1 = xs_row[32 + kk];
            a[0][j] = (_Float16)xd0;
            a[1][j] = (_Float16)xd1;
            a[2][j] = (_Float16)(xs0 - xd0);
            a[3][j] = (_Float16)(xs1 - xd1);
        }
        __syncthreads();   // sDst/sInv ready; orders sHid reuse across iters

        // ---- GEMM1: [16,128] @ [128,64] ----
        v8f acc[4];
#pragma unroll
        for (int nt = 0; nt < 4; ++nt) acc[nt] = vzero;
#pragma unroll
        for (int nt = 0; nt < 4; ++nt) {
#pragma unroll
            for (int c = 0; c < 4; ++c) {
                v16h b = *(const v16h*)(sW1f + (((c * 4 + nt) * 32) + lane) * 16);
                acc[nt] = __builtin_amdgcn_wmma_f32_16x16x32_f16(
                    false, a[c], false, b, (short)0, acc[nt], false, false);
            }
        }

        // bias + relu, stage hidden [row][k] for GEMM2 A-fragments
#pragma unroll
        for (int nt = 0; nt < 4; ++nt) {
            float bias = sB1[16 * nt + m];
#pragma unroll
            for (int r = 0; r < 8; ++r) {
                float h = fmaxf(acc[nt][r] + bias, 0.0f);
                sHid[wave][(r + 8 * g) * HDIM + 16 * nt + m] = (_Float16)h;
            }
        }
        __syncthreads();

        // ---- GEMM2: [16,64] @ [64,64] ----
        v16h h2[2];
#pragma unroll
        for (int j = 0; j < 16; ++j) {
            int kk = (j < 8) ? (8 * g + j) : (16 + 8 * g + (j - 8));
            h2[0][j] = sHid[wave][m * HDIM + kk];
            h2[1][j] = sHid[wave][m * HDIM + 32 + kk];
        }
        v8f acc2[4];
#pragma unroll
        for (int nt = 0; nt < 4; ++nt) acc2[nt] = vzero;
#pragma unroll
        for (int nt = 0; nt < 4; ++nt) {
#pragma unroll
            for (int c = 0; c < 2; ++c) {
                v16h b = *(const v16h*)(sW2f + (((c * 4 + nt) * 32) + lane) * 16);
                acc2[nt] = __builtin_amdgcn_wmma_f32_16x16x32_f16(
                    false, h2[c], false, b, (short)0, acc2[nt], false, false);
            }
        }

        // bias + relu + inv_deg pre-scale + scatter-add (== segment_sum / deg)
#pragma unroll
        for (int r = 0; r < 8; ++r) {
            int row = r + 8 * g;
            int eo  = e0 + row;
            if (eo < E && e0 >= 0) {
                int   dn = sDst[wave][row];
                float sc = sInv[wave][row];
#pragma unroll
                for (int nt = 0; nt < 4; ++nt) {
                    float v = fmaxf(acc2[nt][r] + sB2[16 * nt + m], 0.0f) * sc;
                    atomicAdd(&x_out[(long long)dn * DDIM + 16 * nt + m], v);
                }
            }
        }
    }
}

// ---------------------------------------------------------------- head ------
// logits = relu(x @ W0 + b0) @ Wc + bc
__global__ void head_kernel(const float* __restrict__ x,
                            const float* __restrict__ W0,
                            const float* __restrict__ b0,
                            const float* __restrict__ Wc,
                            const float* __restrict__ bc,
                            float* __restrict__ out, int n) {
    __shared__ float sW0[DDIM * DDIM];
    __shared__ float sWc[DDIM * CDIM];
    __shared__ float sB0[DDIM], sBc[CDIM];
    for (int t = threadIdx.x; t < DDIM * DDIM; t += blockDim.x) sW0[t] = W0[t];
    for (int t = threadIdx.x; t < DDIM * CDIM; t += blockDim.x) sWc[t] = Wc[t];
    if (threadIdx.x < DDIM) sB0[threadIdx.x] = b0[threadIdx.x];
    if (threadIdx.x < CDIM) sBc[threadIdx.x] = bc[threadIdx.x];
    __syncthreads();
    int i = blockIdx.x * blockDim.x + threadIdx.x;
    if (i >= n) return;
    float xr[DDIM];
#pragma unroll
    for (int k = 0; k < DDIM; ++k) xr[k] = x[(long long)i * DDIM + k];
    float lg[CDIM];
#pragma unroll
    for (int c = 0; c < CDIM; ++c) lg[c] = sBc[c];
    for (int o = 0; o < DDIM; ++o) {
        float h = sB0[o];
#pragma unroll
        for (int k = 0; k < DDIM; ++k) h = fmaf(xr[k], sW0[k * DDIM + o], h);
        h = fmaxf(h, 0.0f);
#pragma unroll
        for (int c = 0; c < CDIM; ++c) lg[c] = fmaf(h, sWc[o * CDIM + c], lg[c]);
    }
#pragma unroll
    for (int c = 0; c < CDIM; ++c) out[(long long)i * CDIM + c] = lg[c];
}

// ------------------------------------------------------------- launcher -----
extern "C" void kernel_launch(void* const* d_in, const int* in_sizes, int n_in,
                              void* d_out, int out_size, void* d_ws, size_t ws_size,
                              hipStream_t stream) {
    const float*     nf   = (const float*)d_in[0];
    const long long* ei   = (const long long*)d_in[1];
    // d_in[2] = object_size: unused (segments partition nodes; head is pointwise)
    const float* encW = (const float*)d_in[3];
    const float* encB = (const float*)d_in[4];
    const float* c0W1 = (const float*)d_in[5];
    const float* c0b1 = (const float*)d_in[6];
    const float* c0W2 = (const float*)d_in[7];
    const float* c0b2 = (const float*)d_in[8];
    const float* c1W1 = (const float*)d_in[9];
    const float* c1b1 = (const float*)d_in[10];
    const float* c1W2 = (const float*)d_in[11];
    const float* c1b2 = (const float*)d_in[12];
    const float* pW0  = (const float*)d_in[13];
    const float* pb0  = (const float*)d_in[14];
    const float* pWc  = (const float*)d_in[15];
    const float* pbc  = (const float*)d_in[16];
    float* out = (float*)d_out;

    const int N = in_sizes[0] / FDIM;   // 60000
    const int E = in_sizes[1] / 2;      // 960000

    float* x0  = (float*)d_ws;                  // [N, 64]
    float* x1  = x0 + (long long)N * DDIM;      // [N, 64]
    float* x2  = x1 + (long long)N * DDIM;      // [N, 64]
    float* deg = x2 + (long long)N * DDIM;      // [N] -> becomes inv_deg

    // zero x1, x2, deg (contiguous)
    long long zn = (long long)N * DDIM * 2 + N;
    zero_kernel<<<2048, 256, 0, stream>>>(x1, zn);

    encoder_kernel<<<(N + 255) / 256, 256, 0, stream>>>(nf, encW, encB, x0, N);
    degree_kernel<<<(E + 255) / 256, 256, 0, stream>>>(ei, deg, E);
    invdeg_kernel<<<(N + 255) / 256, 256, 0, stream>>>(deg, N);

    // Persistent conv blocks: enough to fill the machine, each wave loops
    // over many 16-edge tiles so the 48KB weight staging is amortized.
    const int tiles = (E + 15) / 16;
    int convBlocks = (tiles + 7) / 8;
    if (convBlocks > 1024) convBlocks = 1024;
    conv_kernel<<<convBlocks, 256, 0, stream>>>(x0, x1, ei, deg,
                                                c0W1, c0b1, c0W2, c0b2, E);
    conv_kernel<<<convBlocks, 256, 0, stream>>>(x1, x2, ei, deg,
                                                c1W1, c1b1, c1W2, c1b2, E);

    head_kernel<<<(N + 255) / 256, 256, 0, stream>>>(x2, pW0, pb0, pWc, pbc, out, N);
}
